// STGRF_BILSTM_nosub_24275155157312
// MI455X (gfx1250) — compile-verified, matrix operand order checked by reading, not run on hardware
//
#include <hip/hip_runtime.h>
#include <hip/hip_bf16.h>

typedef __attribute__((ext_vector_type(16))) __bf16 v16bf;
typedef __attribute__((ext_vector_type(8)))  float  v8f;

// Problem constants
constexpr int BB   = 16;
constexpr int NN   = 512;
constexpr int EEn  = 16384;
constexpr int HISTn= 24;
constexpr int PREDn= 12;
constexpr int HIDN = 512;
constexpr int G4n  = 2048;      // 4*HID
constexpr int NTn  = 128;       // G4/16 column tiles
constexpr int KT_H = 16;        // HID/32 k-tiles
constexpr int KT_X1= 32;        // 1024/32 k-tiles

// Packed fragment sizes (elements)
constexpr size_t WHH_E  = (size_t)NTn * KT_H  * 512;   // per dir
constexpr size_t WIH1_E = (size_t)NTn * KT_X1 * 512;   // per dir
constexpr size_t YS0_E  = (size_t)NN * 32 * 512;       // layer0 output, packed A-frags

// Workspace byte offsets
constexpr size_t OFF_WHH  = 0;
constexpr size_t OFF_WIH1 = OFF_WHH  + 4 * WHH_E  * 2;
constexpr size_t OFF_YS0  = OFF_WIH1 + 2 * WIH1_E * 2;
constexpr size_t OFF_BIAS = OFF_YS0  + YS0_E * 2;
constexpr size_t OFF_EAN  = OFF_BIAS + (size_t)4 * 2048 * 4;
constexpr size_t OFF_X    = OFF_EAN  + (size_t)EEn * 2 * 4;
constexpr size_t OFF_X2   = OFF_X    + (size_t)BB * NN * 12 * 4;
constexpr size_t OFF_AGG  = OFF_X2   + (size_t)BB * NN * 21 * 4;
constexpr size_t OFF_XN   = OFF_AGG  + (size_t)BB * NN * 41 * 4;
constexpr size_t OFF_HST  = OFF_XN   + (size_t)BB * NN * 4;
constexpr size_t OFF_CST  = OFF_HST  + (size_t)4 * BB * HIDN * 4;
constexpr size_t OFF_OCAT = OFF_CST  + (size_t)4 * BB * HIDN * 4;

__device__ __forceinline__ float sigm(float x) { return 1.0f / (1.0f + expf(-x)); }

// K-offset of fragment element j within a 32-wide k-tile for a given lane
// (16-bit A/B matrix striping per CDNA5 ISA 7.12.2)
__device__ __forceinline__ int frag_k(int lane, int j) {
    int v = j >> 1, p = j & 1;
    return ((v >> 2) << 4) + ((lane >> 4) << 3) + ((v & 3) << 1) + p;
}

// ---------------- prep kernels ----------------

__global__ void k_init(const float* rain, const float* h0, const float* c0,
                       float* xn, float* hst, float* cst) {
    int gid = blockIdx.x * blockDim.x + threadIdx.x;
    if (gid < BB * NN) {
        int b = gid / NN, n = gid % NN;
        xn[gid] = rain[((size_t)b * HISTn + (HISTn - 1)) * NN + n];
    }
    if (gid < 4 * BB * HIDN) { hst[gid] = h0[gid]; cst[gid] = c0[gid]; }
}

__global__ void k_edge_stats(const float* ea, float* ean) {
    __shared__ float red[1024];
    __shared__ float mu0, mu1, is0, is1;
    int tid = threadIdx.x;
    float s0 = 0, s1 = 0, q0 = 0, q1 = 0;
    for (int e = tid; e < EEn; e += 256) {
        float a = ea[e * 2], b = ea[e * 2 + 1];
        s0 += a; s1 += b; q0 += a * a; q1 += b * b;
    }
    red[tid] = s0; red[256 + tid] = s1; red[512 + tid] = q0; red[768 + tid] = q1;
    __syncthreads();
    for (int st = 128; st; st >>= 1) {
        if (tid < st) {
            red[tid] += red[tid + st];             red[256 + tid] += red[256 + tid + st];
            red[512 + tid] += red[512 + tid + st]; red[768 + tid] += red[768 + tid + st];
        }
        __syncthreads();
    }
    if (tid == 0) {
        mu0 = red[0] / EEn; mu1 = red[256] / EEn;
        float v0 = (red[512] - EEn * mu0 * mu0) / (EEn - 1);
        float v1 = (red[768] - EEn * mu1 * mu1) / (EEn - 1);
        is0 = rsqrtf(v0); is1 = rsqrtf(v1);
    }
    __syncthreads();
    for (int e = tid; e < EEn; e += 256) {
        ean[e * 2]     = (ea[e * 2]     - mu0) * is0;
        ean[e * 2 + 1] = (ea[e * 2 + 1] - mu1) * is1;
    }
}

// Repack a (2048 x K) f32 weight matrix into bf16 WMMA B-fragments:
// dst layout [nt][kt][lane][16], so each lane reads 32 contiguous bytes.
__global__ void k_pack_w(const float* w, __bf16* dst, int K) {
    int gid = blockIdx.x * blockDim.x + threadIdx.x;
    int KT = K >> 5;
    int tot = NTn * KT * 512;
    if (gid >= tot) return;
    int nt = gid / (KT * 512);
    int r  = gid % (KT * 512);
    int kt = r >> 9;
    int e  = r & 511;
    int ln = e >> 4, j = e & 15;
    int nn = ln & 15;
    int kk = frag_k(ln, j);
    dst[gid] = (__bf16)w[(size_t)(nt * 16 + nn) * K + kt * 32 + kk];
}

__global__ void k_bias2(const float* a, const float* b, float* o, int n) {
    int gid = blockIdx.x * blockDim.x + threadIdx.x;
    if (gid < n) o[gid] = a[gid] + b[gid];
}

// ---------------- per-step kernels ----------------

__global__ void k_build_x(const float* xn, const float* feature, float* x, int t) {
    int gid = blockIdx.x * blockDim.x + threadIdx.x;
    if (gid >= BB * NN) return;
    int b = gid / NN, n = gid % NN;
    float* xo = x + (size_t)gid * 12;
    xo[0] = xn[gid];
    const float* f = feature + (((size_t)b * (HISTn + PREDn) + (HISTn + t)) * NN + n) * 11;
    for (int c = 0; c < 11; ++c) xo[1 + c] = f[c];
}

__global__ void k_gnn_edge(const float* x, const int* eidx, const float* ea, const float* ean,
                           const float* wmean, const float* wstd,
                           const float* ew1, const float* eb1,
                           const float* ew2, const float* eb2, float* agg) {
    int e = blockIdx.x * blockDim.x + threadIdx.x;
    int b = blockIdx.y;
    if (e >= EEn) return;
    int sIdx = eidx[e], tIdx = eidx[EEn + e];
    const float* xs = x + ((size_t)b * NN + sIdx) * 12;
    const float* xt = x + ((size_t)b * NN + tIdx) * 12;
    float feat[27];
    for (int c = 0; c < 12; ++c) { feat[c] = xs[c]; feat[12 + c] = xt[c]; }
    float speed = feat[10] * wstd[0] + wmean[0];
    float direc = feat[11] * wstd[1] + wmean[1];
    float dist = ea[e * 2], cdir = ea[e * 2 + 1];
    float theta = fabsf(cdir - direc);
    float ew = 3.0f * speed * cosf(theta * 22.5f) / dist;
    feat[24] = ean[e * 2];
    feat[25] = ean[e * 2 + 1];
    feat[26] = ew > 0.0f ? ew : 0.0f;
    float h1[64];
    for (int o = 0; o < 64; ++o) {
        float acc = eb1[o];
        const float* wr = ew1 + o * 27;
        for (int c = 0; c < 27; ++c) acc += feat[c] * wr[c];
        h1[o] = sigm(acc);
    }
    float* aggp = agg + ((size_t)b * NN + tIdx) * 41;
    for (int o = 0; o < 41; ++o) {
        float acc = eb2[o];
        const float* wr = ew2 + o * 64;
        for (int c = 0; c < 64; ++c) acc += h1[c] * wr[c];
        atomicAdd(&aggp[o], sigm(acc));
    }
}

__global__ void k_gnn_node(const float* agg, const float* x,
                           const float* nw, const float* nb, float* x2) {
    int gid = blockIdx.x * blockDim.x + threadIdx.x;
    if (gid >= BB * NN) return;
    const float* ag = agg + (size_t)gid * 41;
    float* xo = x2 + (size_t)gid * 21;
    for (int o = 0; o < 9; ++o) {
        float acc = nb[o];
        const float* wr = nw + o * 41;
        for (int c = 0; c < 41; ++c) acc += ag[c] * wr[c];
        xo[o] = sigm(acc);
    }
    const float* xi = x + (size_t)gid * 12;
    for (int c = 0; c < 12; ++c) xo[9 + c] = xi[c];
}

// ---------------- LSTM layer 0 (persistent, 1 block per direction) ----------------
// kt-outer / gate-inner: one A fragment feeds 4 WMMAs -> small live set, no spills.

__global__ __launch_bounds__(1024) void k_lstm0(const float* x2, const float* wih_f,
                                                const float* wih_r, const __bf16* whhpk,
                                                const float* biasb, float* hstate,
                                                float* cstate, __bf16* ys0) {
    __shared__ __align__(32) __bf16 hA[8192];   // h packed as WMMA A-fragments [kt][lane][16]
    __shared__ float cS[8192];                  // c, [m*512 + hcol]
    __shared__ float xt[16 * 21];
    int dir  = blockIdx.x;
    int tid  = threadIdx.x, lane = tid & 31, w = tid >> 5;
    const float*  wih  = dir ? wih_r : wih_f;
    const __bf16* wpk  = whhpk + (size_t)dir * WHH_E;
    const float*  bias = biasb + dir * G4n;
    float* hs = hstate + (size_t)dir * BB * HIDN;
    float* cs = cstate + (size_t)dir * BB * HIDN;

    for (int i = tid; i < 8192; i += 1024) {
        int kt = i >> 9, rem = i & 511, ln = rem >> 4, j = rem & 15;
        int m = ln & 15, k = kt * 32 + frag_k(ln, j);
        hA[i] = (__bf16)hs[m * HIDN + k];
        cS[i] = cs[i];
    }
    __syncthreads();

    int mbase = (lane >> 4) << 3;
    int hcol  = w * 16 + (lane & 15);
    // B-fragment base pointers for the 4 gate tiles owned by this wave
    const __bf16* wp0 = wpk + (size_t)(w +  0) * (KT_H * 512) + lane * 16;
    const __bf16* wp1 = wpk + (size_t)(w + 32) * (KT_H * 512) + lane * 16;
    const __bf16* wp2 = wpk + (size_t)(w + 64) * (KT_H * 512) + lane * 16;
    const __bf16* wp3 = wpk + (size_t)(w + 96) * (KT_H * 512) + lane * 16;

    for (int ss = 0; ss < NN; ++ss) {
        int s = dir ? (NN - 1 - ss) : ss;
        if (tid < 336) {
            int m = tid / 21, c = tid % 21;
            xt[tid] = x2[((size_t)m * NN + s) * 21 + c];
        }
        __syncthreads();

        v8f acc0 = {0.f,0.f,0.f,0.f,0.f,0.f,0.f,0.f};
        v8f acc1 = acc0, acc2 = acc0, acc3 = acc0;
        for (int kt = 0; kt < KT_H; ++kt) {
            __builtin_prefetch(wp0 + (kt + 2) * 512, 0, 1);
            v16bf av = *(const v16bf*)(hA + kt * 512 + lane * 16);
            v16bf b0 = *(const v16bf*)(wp0 + kt * 512);
            acc0 = __builtin_amdgcn_wmma_f32_16x16x32_bf16(false, av, false, b0, (short)0, acc0, false, false);
            v16bf b1 = *(const v16bf*)(wp1 + kt * 512);
            acc1 = __builtin_amdgcn_wmma_f32_16x16x32_bf16(false, av, false, b1, (short)0, acc1, false, false);
            v16bf b2 = *(const v16bf*)(wp2 + kt * 512);
            acc2 = __builtin_amdgcn_wmma_f32_16x16x32_bf16(false, av, false, b2, (short)0, acc2, false, false);
            v16bf b3 = *(const v16bf*)(wp3 + kt * 512);
            acc3 = __builtin_amdgcn_wmma_f32_16x16x32_bf16(false, av, false, b3, (short)0, acc3, false, false);
        }

        float hnew[8];
        for (int r = 0; r < 8; ++r) {
            int mm = r + mbase;
            float gi = acc0[r] + bias[hcol];
            float gf = acc1[r] + bias[512 + hcol];
            float gg = acc2[r] + bias[1024 + hcol];
            float go = acc3[r] + bias[1536 + hcol];
            const float* xr = xt + mm * 21;
            const float* wi = wih + (size_t)hcol * 21;
            const float* wf = wih + (size_t)(512 + hcol) * 21;
            const float* wg = wih + (size_t)(1024 + hcol) * 21;
            const float* wo = wih + (size_t)(1536 + hcol) * 21;
            for (int c = 0; c < 21; ++c) {
                float xv = xr[c];
                gi += xv * wi[c]; gf += xv * wf[c];
                gg += xv * wg[c]; go += xv * wo[c];
            }
            float cold = cS[mm * HIDN + hcol];
            float cnew = sigm(gf) * cold + sigm(gi) * tanhf(gg);
            cS[mm * HIDN + hcol] = cnew;
            hnew[r] = sigm(go) * tanhf(cnew);
        }
        __syncthreads();
        for (int r = 0; r < 8; ++r) {
            int mm = r + mbase;
            int kt = hcol >> 5, kk = hcol & 31;
            int ln = mm + (((kk >> 3) & 1) << 4);
            int v  = ((kk >> 4) << 2) + ((kk & 7) >> 1);
            int j  = (v << 1) + (kk & 1);
            __bf16 hb = (__bf16)hnew[r];
            hA[kt * 512 + ln * 16 + j] = hb;
            int ktg = dir * 16 + kt;                 // fwd -> kt 0..15, rev -> 16..31
            ys0[(size_t)s * 16384 + (size_t)ktg * 512 + ln * 16 + j] = hb;
        }
        __syncthreads();
    }

    for (int i = tid; i < 8192; i += 1024) {
        int kt = i >> 9, rem = i & 511, ln = rem >> 4, j = rem & 15;
        int m = ln & 15, k = kt * 32 + frag_k(ln, j);
        hs[m * HIDN + k] = (float)hA[i];
        cs[i] = cS[i];
    }
}

// ---------------- LSTM layer 1 (K = 512 hidden + 1024 input, all WMMA) ----------------

__global__ __launch_bounds__(1024) void k_lstm1(const __bf16* ys0, const __bf16* whhpk2,
                                                const __bf16* wih1pk, const float* biasb2,
                                                float* hstate2, float* cstate2, float* ocat) {
    __shared__ __align__(32) __bf16 hA[8192];
    __shared__ float cS[8192];
    int dir  = blockIdx.x;
    int tid  = threadIdx.x, lane = tid & 31, w = tid >> 5;
    const __bf16* wph  = whhpk2 + (size_t)dir * WHH_E;
    const __bf16* wpx  = wih1pk + (size_t)dir * WIH1_E;
    const float*  bias = biasb2 + dir * G4n;
    float* hs = hstate2 + (size_t)dir * BB * HIDN;
    float* cs = cstate2 + (size_t)dir * BB * HIDN;

    for (int i = tid; i < 8192; i += 1024) {
        int kt = i >> 9, rem = i & 511, ln = rem >> 4, j = rem & 15;
        int m = ln & 15, k = kt * 32 + frag_k(ln, j);
        hA[i] = (__bf16)hs[m * HIDN + k];
        cS[i] = cs[i];
    }
    __syncthreads();

    int mbase = (lane >> 4) << 3;
    int hcol  = w * 16 + (lane & 15);
    const __bf16* wh0 = wph + (size_t)(w +  0) * (KT_H * 512) + lane * 16;
    const __bf16* wh1 = wph + (size_t)(w + 32) * (KT_H * 512) + lane * 16;
    const __bf16* wh2 = wph + (size_t)(w + 64) * (KT_H * 512) + lane * 16;
    const __bf16* wh3 = wph + (size_t)(w + 96) * (KT_H * 512) + lane * 16;
    const __bf16* wx0 = wpx + (size_t)(w +  0) * (KT_X1 * 512) + lane * 16;
    const __bf16* wx1 = wpx + (size_t)(w + 32) * (KT_X1 * 512) + lane * 16;
    const __bf16* wx2 = wpx + (size_t)(w + 64) * (KT_X1 * 512) + lane * 16;
    const __bf16* wx3 = wpx + (size_t)(w + 96) * (KT_X1 * 512) + lane * 16;

    for (int ss = 0; ss < NN; ++ss) {
        int s = dir ? (NN - 1 - ss) : ss;
        const __bf16* ax = ys0 + (size_t)s * 16384 + lane * 16;

        v8f acc0 = {0.f,0.f,0.f,0.f,0.f,0.f,0.f,0.f};
        v8f acc1 = acc0, acc2 = acc0, acc3 = acc0;
        for (int kt = 0; kt < KT_H; ++kt) {              // recurrent part (A from LDS)
            __builtin_prefetch(wh0 + (kt + 2) * 512, 0, 1);
            v16bf av = *(const v16bf*)(hA + kt * 512 + lane * 16);
            v16bf b0 = *(const v16bf*)(wh0 + kt * 512);
            acc0 = __builtin_amdgcn_wmma_f32_16x16x32_bf16(false, av, false, b0, (short)0, acc0, false, false);
            v16bf b1 = *(const v16bf*)(wh1 + kt * 512);
            acc1 = __builtin_amdgcn_wmma_f32_16x16x32_bf16(false, av, false, b1, (short)0, acc1, false, false);
            v16bf b2 = *(const v16bf*)(wh2 + kt * 512);
            acc2 = __builtin_amdgcn_wmma_f32_16x16x32_bf16(false, av, false, b2, (short)0, acc2, false, false);
            v16bf b3 = *(const v16bf*)(wh3 + kt * 512);
            acc3 = __builtin_amdgcn_wmma_f32_16x16x32_bf16(false, av, false, b3, (short)0, acc3, false, false);
        }
        for (int kt = 0; kt < KT_X1; ++kt) {             // layer-0 output part (A from global)
            __builtin_prefetch(wx0 + (kt + 2) * 512, 0, 1);
            v16bf av = *(const v16bf*)(ax + kt * 512);
            v16bf b0 = *(const v16bf*)(wx0 + kt * 512);
            acc0 = __builtin_amdgcn_wmma_f32_16x16x32_bf16(false, av, false, b0, (short)0, acc0, false, false);
            v16bf b1 = *(const v16bf*)(wx1 + kt * 512);
            acc1 = __builtin_amdgcn_wmma_f32_16x16x32_bf16(false, av, false, b1, (short)0, acc1, false, false);
            v16bf b2 = *(const v16bf*)(wx2 + kt * 512);
            acc2 = __builtin_amdgcn_wmma_f32_16x16x32_bf16(false, av, false, b2, (short)0, acc2, false, false);
            v16bf b3 = *(const v16bf*)(wx3 + kt * 512);
            acc3 = __builtin_amdgcn_wmma_f32_16x16x32_bf16(false, av, false, b3, (short)0, acc3, false, false);
        }

        float hnew[8];
        for (int r = 0; r < 8; ++r) {
            int mm = r + mbase;
            float gi = acc0[r] + bias[hcol];
            float gf = acc1[r] + bias[512 + hcol];
            float gg = acc2[r] + bias[1024 + hcol];
            float go = acc3[r] + bias[1536 + hcol];
            float cold = cS[mm * HIDN + hcol];
            float cnew = sigm(gf) * cold + sigm(gi) * tanhf(gg);
            cS[mm * HIDN + hcol] = cnew;
            float hv = sigm(go) * tanhf(cnew);
            hnew[r] = hv;
            ocat[(size_t)s * 16384 + (size_t)mm * 1024 + dir * 512 + hcol] = hv;
        }
        __syncthreads();
        for (int r = 0; r < 8; ++r) {
            int mm = r + mbase;
            int kt = hcol >> 5, kk = hcol & 31;
            int ln = mm + (((kk >> 3) & 1) << 4);
            int v  = ((kk >> 4) << 2) + ((kk & 7) >> 1);
            int j  = (v << 1) + (kk & 1);
            hA[kt * 512 + ln * 16 + j] = (__bf16)hnew[r];
        }
        __syncthreads();
    }

    for (int i = tid; i < 8192; i += 1024) {
        int kt = i >> 9, rem = i & 511, ln = rem >> 4, j = rem & 15;
        int m = ln & 15, k = kt * 32 + frag_k(ln, j);
        hs[m * HIDN + k] = (float)hA[i];
        cs[i] = cS[i];
    }
}

__global__ void k_fc(const float* ocat, const float* fcw, const float* fcb,
                     float* xn, float* out, int t) {
    int gid = blockIdx.x * blockDim.x + threadIdx.x;
    if (gid >= BB * NN) return;
    int b = gid / NN, n = gid % NN;
    const float* oc = ocat + ((size_t)n * 16 + b) * 1024;
    float acc = fcb[0];
    for (int k = 0; k < 1024; ++k) acc += oc[k] * fcw[k];
    xn[gid] = acc;
    out[((size_t)b * PREDn + t) * NN + n] = acc;
}

// ---------------- host ----------------

extern "C" void kernel_launch(void* const* d_in, const int* in_sizes, int n_in,
                              void* d_out, int out_size, void* d_ws, size_t ws_size,
                              hipStream_t stream) {
    const float* rain      = (const float*)d_in[0];
    const float* feature   = (const float*)d_in[1];
    const float* h0        = (const float*)d_in[2];
    const float* c0        = (const float*)d_in[3];
    const float* edge_attr = (const float*)d_in[4];
    const float* wind_mean = (const float*)d_in[5];
    const float* wind_std  = (const float*)d_in[6];
    const float* ew1 = (const float*)d_in[7];
    const float* eb1 = (const float*)d_in[8];
    const float* ew2 = (const float*)d_in[9];
    const float* eb2 = (const float*)d_in[10];
    const float* nw  = (const float*)d_in[11];
    const float* nb  = (const float*)d_in[12];
    const float* fcw = (const float*)d_in[29];
    const float* fcb = (const float*)d_in[30];
    const int*   eidx= (const int*)d_in[31];

    char* ws = (char*)d_ws;
    __bf16* whhpk  = (__bf16*)(ws + OFF_WHH);
    __bf16* wih1pk = (__bf16*)(ws + OFF_WIH1);
    __bf16* ys0    = (__bf16*)(ws + OFF_YS0);
    float*  bias   = (float*)(ws + OFF_BIAS);
    float*  ean    = (float*)(ws + OFF_EAN);
    float*  x      = (float*)(ws + OFF_X);
    float*  x2     = (float*)(ws + OFF_X2);
    float*  agg    = (float*)(ws + OFF_AGG);
    float*  xn     = (float*)(ws + OFF_XN);
    float*  hst    = (float*)(ws + OFF_HST);
    float*  cst    = (float*)(ws + OFF_CST);
    float*  ocat   = (float*)(ws + OFF_OCAT);

    // one-time prep (re-done every call: deterministic)
    k_init<<<128, 256, 0, stream>>>(rain, h0, c0, xn, hst, cst);
    k_edge_stats<<<1, 256, 0, stream>>>(edge_attr, ean);
    const int whh_idx[4] = {14, 18, 22, 26};
    for (int d = 0; d < 4; ++d)
        k_pack_w<<<(int)((WHH_E + 255) / 256), 256, 0, stream>>>(
            (const float*)d_in[whh_idx[d]], whhpk + d * WHH_E, 512);
    const int wih1_idx[2] = {21, 25};
    for (int d = 0; d < 2; ++d)
        k_pack_w<<<(int)((WIH1_E + 255) / 256), 256, 0, stream>>>(
            (const float*)d_in[wih1_idx[d]], wih1pk + d * WIH1_E, 1024);
    const int bi[4] = {15, 19, 23, 27}, bh[4] = {16, 20, 24, 28};
    for (int d = 0; d < 4; ++d)
        k_bias2<<<8, 256, 0, stream>>>((const float*)d_in[bi[d]],
                                       (const float*)d_in[bh[d]], bias + d * G4n, G4n);

    for (int t = 0; t < PREDn; ++t) {
        k_build_x<<<32, 256, 0, stream>>>(xn, feature, x, t);
        hipMemsetAsync(agg, 0, (size_t)BB * NN * 41 * 4, stream);
        k_gnn_edge<<<dim3(EEn / 256, BB), 256, 0, stream>>>(
            x, eidx, edge_attr, ean, wind_mean, wind_std, ew1, eb1, ew2, eb2, agg);
        k_gnn_node<<<32, 256, 0, stream>>>(agg, x, nw, nb, x2);
        k_lstm0<<<2, 1024, 0, stream>>>(x2, (const float*)d_in[13], (const float*)d_in[17],
                                        whhpk, bias, hst, cst, ys0);
        k_lstm1<<<2, 1024, 0, stream>>>(ys0, whhpk + 2 * WHH_E, wih1pk, bias + 2 * G4n,
                                        hst + 2 * BB * HIDN, cst + 2 * BB * HIDN, ocat);
        k_fc<<<32, 256, 0, stream>>>(ocat, fcw, fcb, xn, (float*)d_out, t);
    }
}
